// GCN_19791209300127
// MI455X (gfx1250) — matrix operand order repaired and checked by reading, not compiled, not run on hardware
//
#include <hip/hip_runtime.h>
#include <hip/hip_bf16.h>
#include <math.h>

typedef __bf16 bf16;
typedef __attribute__((ext_vector_type(16))) __bf16 v16bf;
typedef __attribute__((ext_vector_type(8)))  __bf16 v8bf;
typedef __attribute__((ext_vector_type(8)))  float  v8f;
typedef unsigned int u32x4 __attribute__((ext_vector_type(4)));
typedef int          i32x4 __attribute__((ext_vector_type(4)));
typedef int          i32x8 __attribute__((ext_vector_type(8)));

static constexpr int Nn   = 4096;   // nodes
static constexpr int Ff   = 1024;   // features
static constexpr int Mm   = 4096;   // hyperedges (edge_attr rows)
static constexpr int HIDD = 512;
static constexpr int FH2  = 2048;   // F + 2*HID

// ---------------------------------------------------------------- utilities

__global__ void k_fill(float* p, float v, size_t n) {
  size_t i = (size_t)blockIdx.x * 256 + threadIdx.x;
  if (i < n) p[i] = v;
}

__global__ void k_cvt(bf16* o, const float* in, size_t n) {
  size_t i = (size_t)blockIdx.x * 256 + threadIdx.x;
  if (i < n) o[i] = (bf16)in[i];
}

// in: [K][N] f32 row-major  ->  o: [N][K] bf16 row-major (i.e. B^T)
__global__ void k_cvtT(bf16* __restrict__ o, const float* __restrict__ in, int K, int N) {
  __shared__ float tile[32][33];
  int kb = blockIdx.y * 32, nb = blockIdx.x * 32;
  int tx = threadIdx.x & 31, ty = threadIdx.x >> 5;   // 256 threads = 32x8
  for (int r = ty; r < 32; r += 8) {
    int k = kb + r, n = nb + tx;
    tile[r][tx] = (k < K && n < N) ? in[(size_t)k * N + n] : 0.f;
  }
  __syncthreads();
  for (int r = ty; r < 32; r += 8) {
    int n = nb + r, k = kb + tx;
    if (n < N && k < K) o[(size_t)n * K + k] = (bf16)tile[tx][r];
  }
}

// ---------------------------------------------------------------- WMMA GEMM + TDM

__device__ inline v16bf load_afrag(const bf16* p) {
  v8bf lo = *(const v8bf*)p;
  v8bf hi = *(const v8bf*)(p + 16);
  return __builtin_shufflevector(lo, hi, 0,1,2,3,4,5,6,7,8,9,10,11,12,13,14,15);
}
__device__ inline v16bf load_bfrag_lds(const bf16* p) {
  v8bf lo = *(const v8bf*)p;
  v8bf hi = *(const v8bf*)(p + 16);
  return __builtin_shufflevector(lo, hi, 0,1,2,3,4,5,6,7,8,9,10,11,12,13,14,15);
}

// TDM: DMA a 64(n-rows) x 64(k) bf16 tile of Bt into LDS, 16B pad per 128B row
// (LDS row stride = 144B = 72 elements). Descriptor per CDNA5 ISA sec. 8.
__device__ inline void tdm_load_B(const bf16* Bt, int n0, int K, int N, int kc, unsigned ldsoff) {
  unsigned long long ga = (unsigned long long)(size_t)(Bt + (size_t)n0 * K + kc);
  u32x4 g0;
  g0[0] = 1u;                                            // count=1 (valid user D#)
  g0[1] = ldsoff;                                        // lds_addr (bytes)
  g0[2] = (unsigned)ga;                                  // global_addr[31:0]
  g0[3] = ((unsigned)(ga >> 32) & 0x1ffffffu) | (2u << 30); // global_addr[56:32], type=2
  i32x8 g1;
  g1[0] = (1 << 16) | (1 << 20) | (4 << 22) | (3 << 25); // data_size=2B, pad_en, 32DW interval, 4DW pad
  g1[1] = (int)(((unsigned)K & 0xffffu) << 16);          // tensor_dim0[15:0] (barrier addr = 0)
  g1[2] = (int)(((unsigned)K >> 16) | (((unsigned)N & 0xffffu) << 16)); // td0 hi | tensor_dim1 lo
  g1[3] = (int)(((unsigned)N >> 16) | (64u << 16));      // td1 hi | tile_dim0=64
  g1[4] = 64;                                            // tile_dim1=64, tile_dim2=0
  g1[5] = K;                                             // tensor_dim0_stride[31:0]
  g1[6] = 0;                                             // stride0 hi | stride1 lo
  g1[7] = 0;
  i32x4 z4 = {0, 0, 0, 0};
#if __clang_major__ >= 23
  i32x8 z8 = {0, 0, 0, 0, 0, 0, 0, 0};
  __builtin_amdgcn_tensor_load_to_lds(g0, g1, z4, z4, z8, 0);
#else
  __builtin_amdgcn_tensor_load_to_lds(g0, g1, z4, z4, 0);
#endif
}

// C[M,N] = act( A_bf16[M,K] @ Bt_bf16[N,K]^T + bias + res ), optional bf16 copy.
// Block: 8 waves, 256(M) x 64(N); wave tile 32x64 (8 accumulators, A reused 4x).
// B staged through LDS by the Tensor Data Mover, double buffered, K-chunk = 64.
// __launch_bounds__(256, 2): target 2 waves/SIMD so the allocator keeps all four
// B fragments resident (removes per-fragment s_wait_dscnt serialization).
// act: 0=none, 1=leaky(0.01), 2=relu. Requires M%256==0, N%64==0, K%64==0.
__global__ void __launch_bounds__(256, 2)
k_gemm_bf16(float* __restrict__ C, bf16* __restrict__ Cb,
            const bf16* __restrict__ A, const bf16* __restrict__ Bt,
            const float* __restrict__ bias, const float* __restrict__ res,
            int M, int N, int K, int act) {
  __shared__ __align__(32) bf16 sB[2][64 * 72];          // 2 x 9216B, padded rows
  const int lane = threadIdx.x & 31;
  const int wave = threadIdx.x >> 5;
  const int n0 = blockIdx.x * 64;
  const int m0 = blockIdx.y * 256 + wave * 32;
  const int lh = lane & 15;        // m (A) / n (B,C)
  const int lg = lane >> 4;        // K-half selector

  const unsigned lds0 = (unsigned)(size_t)(void*)&sB[0][0];
  const unsigned lds1 = (unsigned)(size_t)(void*)&sB[1][0];

  v8f acc[8];
#pragma unroll
  for (int i = 0; i < 8; ++i) acc[i] = (v8f){};

  const bf16* ap0 = A + (size_t)(m0 + lh) * K + lg * 8;        // rows m0..m0+15
  const bf16* ap1 = A + (size_t)(m0 + 16 + lh) * K + lg * 8;   // rows m0+16..m0+31
  const bf16* bl = &sB[0][0] + (size_t)lh * 72 + lg * 8;       // frag base (t=0, buf 0)

  if (wave == 0) tdm_load_B(Bt, n0, K, N, 0, lds0);

  for (int kc = 0; kc < K; kc += 64) {
    const int cur = (kc >> 6) & 1;
    if (wave == 0) {
      if (kc + 64 < K) {
        tdm_load_B(Bt, n0, K, N, kc + 64, cur ? lds0 : lds1);
        __builtin_amdgcn_s_wait_tensorcnt(1);   // current chunk landed in LDS
      } else {
        __builtin_amdgcn_s_wait_tensorcnt(0);
      }
    }
    __syncthreads();                            // publish tile to all waves

    const bf16* bc = bl + (size_t)cur * (64 * 72);
#pragma unroll
    for (int kk2 = 0; kk2 < 2; ++kk2) {         // two 32-K WMMA steps per chunk
      const int kk = kk2 * 32;
      // All fragments into distinct registers first: one ds clause + one wait,
      // then 8 back-to-back WMMAs (no per-fragment s_wait_dscnt stalls).
      v16bf a0  = load_afrag(ap0 + kc + kk);
      v16bf a1  = load_afrag(ap1 + kc + kk);
      v16bf bb0 = load_bfrag_lds(bc + 0 * (16 * 72) + kk);
      v16bf bb1 = load_bfrag_lds(bc + 1 * (16 * 72) + kk);
      v16bf bb2 = load_bfrag_lds(bc + 2 * (16 * 72) + kk);
      v16bf bb3 = load_bfrag_lds(bc + 3 * (16 * 72) + kk);
      acc[0] = __builtin_amdgcn_wmma_f32_16x16x32_bf16(false, a0, false, bb0, (short)0, acc[0], false, false);
      acc[4] = __builtin_amdgcn_wmma_f32_16x16x32_bf16(false, a1, false, bb0, (short)0, acc[4], false, false);
      acc[1] = __builtin_amdgcn_wmma_f32_16x16x32_bf16(false, a0, false, bb1, (short)0, acc[1], false, false);
      acc[5] = __builtin_amdgcn_wmma_f32_16x16x32_bf16(false, a1, false, bb1, (short)0, acc[5], false, false);
      acc[2] = __builtin_amdgcn_wmma_f32_16x16x32_bf16(false, a0, false, bb2, (short)0, acc[2], false, false);
      acc[6] = __builtin_amdgcn_wmma_f32_16x16x32_bf16(false, a1, false, bb2, (short)0, acc[6], false, false);
      acc[3] = __builtin_amdgcn_wmma_f32_16x16x32_bf16(false, a0, false, bb3, (short)0, acc[3], false, false);
      acc[7] = __builtin_amdgcn_wmma_f32_16x16x32_bf16(false, a1, false, bb3, (short)0, acc[7], false, false);
    }
    __syncthreads();                            // tile consumed; safe to overwrite
  }

#pragma unroll
  for (int h = 0; h < 2; ++h) {
#pragma unroll
    for (int t = 0; t < 4; ++t) {
      const int col = n0 + t * 16 + lh;
      const float bv = bias ? bias[col] : 0.f;
#pragma unroll
      for (int j = 0; j < 8; ++j) {
        const int row = m0 + h * 16 + lg * 8 + j;  // C layout: VGPR j -> M=j / M=8+j
        size_t idx = (size_t)row * N + col;
        float v = acc[h * 4 + t][j] + bv;
        if (res) v += res[idx];
        if (act == 1) v = v > 0.f ? v : 0.01f * v;
        else if (act == 2) v = v > 0.f ? v : 0.f;
        C[idx] = v;
        if (Cb) Cb[idx] = (bf16)v;
      }
    }
  }
}

// ---------------------------------------------------------------- reductions

// out[row] = dot(A[row,:], v) + (bias ? bias[0] : 0)
__global__ void k_rowdot(float* out, const float* __restrict__ A, const float* __restrict__ v,
                         int K, const float* bias) {
  int row = blockIdx.x;
  const float* a = A + (size_t)row * K;
  float s = 0.f;
  for (int k = threadIdx.x; k < K; k += 256) s += a[k] * v[k];
  __shared__ float red[256];
  red[threadIdx.x] = s; __syncthreads();
  for (int o = 128; o > 0; o >>= 1) {
    if (threadIdx.x < o) red[threadIdx.x] += red[threadIdx.x + o];
    __syncthreads();
  }
  if (threadIdx.x == 0) out[row] = red[0] + (bias ? bias[0] : 0.f);
}

// ---------------------------------------------------------------- hyperedge ops

__device__ inline void atomicMaxF(float* addr, float val) {
  unsigned int* ua = (unsigned int*)addr;
  unsigned int old = __float_as_uint(*addr);
  while (__uint_as_float(old) < val) {
    unsigned int assumed = old;
    old = atomicCAS(ua, assumed, __float_as_uint(val));
    if (old == assumed) break;
  }
}

__global__ void k_degree(const int* node, const int* he, float* D, float* B, int E) {
  int e = blockIdx.x * 256 + threadIdx.x;
  if (e >= E) return;
  atomicAdd(&D[node[e]], 1.f);
  atomicAdd(&B[he[e]], 1.f);
}

__global__ void k_recip(float* p, int n) {
  int i = blockIdx.x * 256 + threadIdx.x;
  if (i < n) p[i] = p[i] > 0.f ? 1.f / p[i] : 0.f;
}

__global__ void k_edge_a(const int* node, const int* he, const float* an, const float* ae,
                         float* a_e, float* amax, int E) {
  int e = blockIdx.x * 256 + threadIdx.x;
  if (e >= E) return;
  float a = an[node[e]] + ae[he[e]];
  a = a > 0.f ? a : 0.2f * a;           // leaky_relu(0.2)
  a_e[e] = a;
  atomicMaxF(&amax[he[e]], a);
}

__global__ void k_edge_exp(const int* he, float* a_e, const float* amax, float* asum, int E) {
  int e = blockIdx.x * 256 + threadIdx.x;
  if (e >= E) return;
  int h = he[e];
  float am = amax[h];
  if (!(am > -1e37f && am < 1e37f)) am = 0.f;   // isfinite fallback
  float ex = expf(a_e[e] - am);
  a_e[e] = ex;
  atomicAdd(&asum[h], ex);
}

__global__ void k_edge_w(const int* node, const int* he, const float* a_e, const float* asum,
                         const float* Dinv, const float* Binv, float* wef, float* wout, int E) {
  int e = blockIdx.x * 256 + threadIdx.x;
  if (e >= E) return;
  int h = he[e], nd = node[e];
  float soft = a_e[e] / (asum[h] + 1e-16f);
  wef[e]  = Binv[h] * soft;
  wout[e] = Dinv[nd] * soft;
}

// ef[he[e],:] += wef[e] * xl[node[e],:]
__global__ void k_scatter_ef(const int* node, const int* he, const float* wef,
                             const float* __restrict__ xl, float* ef, int F) {
  int e = blockIdx.x;
  float w = wef[e];
  const float* src = xl + (size_t)node[e] * F;
  float* dst = ef + (size_t)he[e] * F;
  for (int f = threadIdx.x; f < F; f += 256) atomicAdd(&dst[f], w * src[f]);
}

// out[node[e],:] += wout[e] * ef[he[e],:]
__global__ void k_scatter_out(const int* node, const int* he, const float* wout,
                              const float* __restrict__ ef, float* out, int F) {
  int e = blockIdx.x;
  float w = wout[e];
  const float* src = ef + (size_t)he[e] * F;
  float* dst = out + (size_t)node[e] * F;
  for (int f = threadIdx.x; f < F; f += 256) atomicAdd(&dst[f], w * src[f]);
}

__global__ void k_add_bias(float* p, const float* bias, int F, size_t total) {
  size_t i = (size_t)blockIdx.x * 256 + threadIdx.x;
  if (i < total) p[i] += bias[i % (size_t)F];
}

// ---------------------------------------------------------------- GraphNorm

__global__ void k_colsum(const float* __restrict__ h, float* acc, int F, int rowsPerBlk) {
  int f = blockIdx.x * 256 + threadIdx.x;
  int r0 = blockIdx.y * rowsPerBlk;
  float s = 0.f;
  for (int r = r0; r < r0 + rowsPerBlk; ++r) s += h[(size_t)r * F + f];
  atomicAdd(&acc[f], s);
}

__global__ void k_colsq(const float* __restrict__ h, const float* csum, const float* ms,
                        float* acc, int F, int rowsPerBlk, float invN) {
  int f = blockIdx.x * 256 + threadIdx.x;
  int r0 = blockIdx.y * rowsPerBlk;
  float mshift = csum[f] * invN * ms[f];
  float s = 0.f;
  for (int r = r0; r < r0 + rowsPerBlk; ++r) {
    float v = h[(size_t)r * F + f] - mshift;
    s += v * v;
  }
  atomicAdd(&acc[f], s);
}

__global__ void k_gnorm_apply(bf16* __restrict__ ob, const float* __restrict__ h,
                              const float* csum, const float* cvar,
                              const float* w, const float* b, const float* ms,
                              float invN, int F, size_t total) {
  size_t i = (size_t)blockIdx.x * 256 + threadIdx.x;
  if (i >= total) return;
  int f = (int)(i % (size_t)F);
  float mean = csum[f] * invN;
  float var  = cvar[f] * invN;
  float v = (h[i] - mean * ms[f]) * rsqrtf(var + 1e-5f) * w[f] + b[f];
  v = v > 0.f ? v : 0.01f * v;          // fused leaky(0.01)
  ob[i] = (bf16)v;
}

// ---------------------------------------------------------------- head glue

// outT_bf16[i, kn] = out[kn, i], out = concat(x, out1, out2) along features
__global__ void k_build_outT(bf16* o, const float* __restrict__ x, const float* __restrict__ o1,
                             const float* __restrict__ o2, size_t total) {
  size_t idx = (size_t)blockIdx.x * 256 + threadIdx.x;
  if (idx >= total) return;
  int kn = (int)(idx & 4095);
  int i  = (int)(idx >> 12);
  float v;
  if (i < 1024)      v = x [(size_t)kn * 1024 + i];
  else if (i < 1536) v = o1[(size_t)kn * 512 + (i - 1024)];
  else               v = o2[(size_t)kn * 512 + (i - 1536)];
  o[idx] = (bf16)v;
}

__global__ void k_sigmoid(float* p, int n) {
  int i = blockIdx.x * 256 + threadIdx.x;
  if (i < n) p[i] = 1.f / (1.f + expf(-p[i]));
}

__global__ void k_meansub(float* s, int n) {
  __shared__ float red[256];
  float acc = 0.f;
  for (int i = threadIdx.x; i < n; i += 256) acc += s[i];
  red[threadIdx.x] = acc; __syncthreads();
  for (int o = 128; o > 0; o >>= 1) {
    if (threadIdx.x < o) red[threadIdx.x] += red[threadIdx.x + o];
    __syncthreads();
  }
  float mean = red[0] / (float)n;
  for (int i = threadIdx.x; i < n; i += 256) s[i] -= mean;
}

// Hb0_bf16[r, i] = out[r, i] * s[i]   (r < BS)
__global__ void k_build_Hb0(bf16* o, const float* __restrict__ x, const float* __restrict__ o1,
                            const float* __restrict__ o2, const float* __restrict__ s, size_t total) {
  size_t idx = (size_t)blockIdx.x * 256 + threadIdx.x;
  if (idx >= total) return;
  int i = (int)(idx & 2047);
  int r = (int)(idx >> 11);
  float v;
  if (i < 1024)      v = x [(size_t)r * 1024 + i];
  else if (i < 1536) v = o1[(size_t)r * 512 + (i - 1024)];
  else               v = o2[(size_t)r * 512 + (i - 1536)];
  o[idx] = (bf16)(v * s[i]);
}

// out[m, 0..1] = A[m,:] @ W[:,0..1] + b   (W row-major [K,2])
__global__ void k_head(float* out, const float* __restrict__ A, const float* __restrict__ W,
                       const float* b, int K) {
  int m = blockIdx.x;
  const float* a = A + (size_t)m * K;
  float s0 = 0.f, s1 = 0.f;
  for (int k = threadIdx.x; k < K; k += 256) {
    float av = a[k];
    s0 += av * W[2 * k];
    s1 += av * W[2 * k + 1];
  }
  __shared__ float r0[256], r1[256];
  r0[threadIdx.x] = s0; r1[threadIdx.x] = s1; __syncthreads();
  for (int o = 128; o > 0; o >>= 1) {
    if (threadIdx.x < o) { r0[threadIdx.x] += r0[threadIdx.x + o]; r1[threadIdx.x] += r1[threadIdx.x + o]; }
    __syncthreads();
  }
  if (threadIdx.x == 0) { out[2 * m] = r0[0] + b[0]; out[2 * m + 1] = r1[0] + b[1]; }
}

// ================================================================ launch

extern "C" void kernel_launch(void* const* d_in, const int* in_sizes, int n_in,
                              void* d_out, int out_size, void* d_ws, size_t ws_size,
                              hipStream_t stream) {
  const float* x         = (const float*)d_in[0];
  const float* edge_attr = (const float*)d_in[1];
  const float* lin1_w = (const float*)d_in[2];
  const float* att1   = (const float*)d_in[3];
  const float* bias1  = (const float*)d_in[4];
  const float* gn1_w  = (const float*)d_in[5];
  const float* gn1_b  = (const float*)d_in[6];
  const float* gn1_ms = (const float*)d_in[7];
  const float* fc1_w  = (const float*)d_in[8];
  const float* fc1_b  = (const float*)d_in[9];
  const float* lin2_w = (const float*)d_in[10];
  const float* att2   = (const float*)d_in[11];
  const float* bias2  = (const float*)d_in[12];
  const float* gn2_w  = (const float*)d_in[13];
  const float* gn2_b  = (const float*)d_in[14];
  const float* gn2_ms = (const float*)d_in[15];
  const float* fc2_w  = (const float*)d_in[16];
  const float* fc2_b  = (const float*)d_in[17];
  const float* at1_w  = (const float*)d_in[18];
  const float* at1_b  = (const float*)d_in[19];
  const float* at2_w  = (const float*)d_in[20];
  const float* at2_b  = (const float*)d_in[21];
  const float* cf1_w  = (const float*)d_in[22];
  const float* cf1_b  = (const float*)d_in[23];
  const float* dh_w   = (const float*)d_in[24];
  const float* dh_b   = (const float*)d_in[25];
  const float* cf2_w  = (const float*)d_in[26];
  const float* cf2_b  = (const float*)d_in[27];
  const float* ch_w   = (const float*)d_in[28];
  const float* ch_b   = (const float*)d_in[29];
  const int*   ei     = (const int*)d_in[30];
  const int E  = in_sizes[30] / 2;
  const int BS = out_size / 4;          // 2 outputs x OUT=2 per row

  // ---- workspace carve-out
  char* wp = (char*)d_ws;
  auto alloc = [&](size_t bytes) -> void* {
    void* p = (void*)wp;
    wp += (bytes + 255) & ~(size_t)255;
    return p;
  };
  const size_t NF = (size_t)Nn * Ff;
  float* xl   = (float*)alloc(NF * 4);                 // xl1 then xl2
  float* ea   = (float*)alloc(NF * 4);                 // ea1 then ea2
  float* ef   = (float*)alloc((size_t)Mm * Ff * 4);
  float* hbuf = (float*)alloc(NF * 4);                 // hconv output (both passes)
  float* out1 = (float*)alloc((size_t)Nn * HIDD * 4);
  float* out2 = (float*)alloc((size_t)Nn * HIDD * 4);
  float* tmat = (float*)alloc((size_t)FH2 * Nn * 4);
  float* anode = (float*)alloc(Nn * 4);
  float* aedge = (float*)alloc(Mm * 4);
  float* a_e  = (float*)alloc((size_t)E * 4);
  float* wef  = (float*)alloc((size_t)E * 4);
  float* wout = (float*)alloc((size_t)E * 4);
  float* amax = (float*)alloc(Mm * 4);
  float* asum = (float*)alloc(Mm * 4);
  float* Dinv = (float*)alloc(Nn * 4);
  float* Binv = (float*)alloc(Mm * 4);
  float* csum = (float*)alloc(Ff * 4);
  float* cvar = (float*)alloc(Ff * 4);
  float* svec = (float*)alloc(FH2 * 4);
  float* Hb1  = (float*)alloc((size_t)BS * HIDD * 4);
  float* Hb2  = (float*)alloc((size_t)BS * HIDD * 4);
  bf16* xb    = (bf16*)alloc(NF * 2);
  bf16* eb    = (bf16*)alloc(NF * 2);
  bf16* hb    = (bf16*)alloc(NF * 2);
  bf16* h2b   = (bf16*)alloc(NF * 2);
  bf16* l1t   = (bf16*)alloc((size_t)Ff * Ff * 2);
  bf16* l2t   = (bf16*)alloc((size_t)Ff * Ff * 2);
  bf16* fc1t  = (bf16*)alloc((size_t)HIDD * Ff * 2);
  bf16* fc2t  = (bf16*)alloc((size_t)HIDD * Ff * 2);
  bf16* at1t  = (bf16*)alloc((size_t)Nn * Nn * 2);
  bf16* outTb = (bf16*)alloc((size_t)FH2 * Nn * 2);
  bf16* cf1t  = (bf16*)alloc((size_t)HIDD * FH2 * 2);
  bf16* cf2t  = (bf16*)alloc((size_t)HIDD * HIDD * 2);
  bf16* Hb0b  = (bf16*)alloc((size_t)BS * FH2 * 2);
  bf16* Hb1b  = (bf16*)alloc((size_t)BS * HIDD * 2);

  const int* nodeI = ei;
  const int* heI   = ei + E;

  auto fill = [&](float* p, float v, size_t n) {
    k_fill<<<dim3((unsigned)((n + 255) / 256)), 256, 0, stream>>>(p, v, n);
  };
  auto gemm = [&](float* C, bf16* Cb, const bf16* A, const bf16* Bt,
                  const float* bias, const float* res, int Mx, int Nx, int Kx, int act) {
    k_gemm_bf16<<<dim3(Nx / 64, (Mx + 255) / 256), 256, 0, stream>>>(C, Cb, A, Bt, bias, res, Mx, Nx, Kx, act);
  };
  auto cvtT = [&](bf16* o, const float* in, int Kx, int Nx) {
    k_cvtT<<<dim3((Nx + 31) / 32, (Kx + 31) / 32), 256, 0, stream>>>(o, in, Kx, Nx);
  };
  const unsigned Eblk = (unsigned)((E + 255) / 256);

  auto hconv_scatter = [&](const float* att, const float* biasv, float* xlp, float* eap, float* outp) {
    k_rowdot<<<Nn, 256, 0, stream>>>(anode, xlp, att, Ff, nullptr);
    k_rowdot<<<Mm, 256, 0, stream>>>(aedge, eap, att + Ff, Ff, nullptr);
    fill(amax, -INFINITY, Mm);
    fill(asum, 0.f, Mm);
    k_edge_a<<<Eblk, 256, 0, stream>>>(nodeI, heI, anode, aedge, a_e, amax, E);
    k_edge_exp<<<Eblk, 256, 0, stream>>>(heI, a_e, amax, asum, E);
    k_edge_w<<<Eblk, 256, 0, stream>>>(nodeI, heI, a_e, asum, Dinv, Binv, wef, wout, E);
    fill(ef, 0.f, (size_t)Mm * Ff);
    k_scatter_ef<<<E, 256, 0, stream>>>(nodeI, heI, wef, xlp, ef, Ff);
    fill(outp, 0.f, NF);
    k_scatter_out<<<E, 256, 0, stream>>>(nodeI, heI, wout, ef, outp, Ff);
    k_add_bias<<<dim3((unsigned)((NF + 255) / 256)), 256, 0, stream>>>(outp, biasv, Ff, NF);
  };
  auto gnorm = [&](const float* inp, const float* gw, const float* gb, const float* gms, bf16* outb) {
    fill(csum, 0.f, Ff);
    k_colsum<<<dim3(Ff / 256, 16), 256, 0, stream>>>(inp, csum, Ff, Nn / 16);
    fill(cvar, 0.f, Ff);
    k_colsq<<<dim3(Ff / 256, 16), 256, 0, stream>>>(inp, csum, gms, cvar, Ff, Nn / 16, 1.f / Nn);
    k_gnorm_apply<<<dim3((unsigned)((NF + 255) / 256)), 256, 0, stream>>>(
        outb, inp, csum, cvar, gw, gb, gms, 1.f / Nn, Ff, NF);
  };

  // ---- one-time conversions (A operands + transposed bf16 weights)
  k_cvt<<<dim3((unsigned)((NF + 255) / 256)), 256, 0, stream>>>(xb, x, NF);
  k_cvt<<<dim3((unsigned)((NF + 255) / 256)), 256, 0, stream>>>(eb, edge_attr, NF);
  cvtT(l1t, lin1_w, Ff, Ff);
  cvtT(l2t, lin2_w, Ff, Ff);
  cvtT(fc1t, fc1_w, Ff, HIDD);
  cvtT(fc2t, fc2_w, Ff, HIDD);
  cvtT(at1t, at1_w, Nn, Nn);
  cvtT(cf1t, cf1_w, FH2, HIDD);
  cvtT(cf2t, cf2_w, HIDD, HIDD);

  // ---- degrees (shared by both conv passes)
  fill(Dinv, 0.f, Nn);
  fill(Binv, 0.f, Mm);
  k_degree<<<Eblk, 256, 0, stream>>>(nodeI, heI, Dinv, Binv, E);
  k_recip<<<dim3((Nn + 255) / 256), 256, 0, stream>>>(Dinv, Nn);
  k_recip<<<dim3((Mm + 255) / 256), 256, 0, stream>>>(Binv, Mm);

  // ---- hconv1 + gnorm1 + fc1
  gemm(xl, nullptr, xb, l1t, nullptr, nullptr, Nn, Ff, Ff, 0);
  gemm(ea, nullptr, eb, l1t, nullptr, nullptr, Mm, Ff, Ff, 0);
  hconv_scatter(att1, bias1, xl, ea, hbuf);
  gnorm(hbuf, gn1_w, gn1_b, gn1_ms, hb);
  gemm(out1, nullptr, hb, fc1t, fc1_b, nullptr, Nn, HIDD, Ff, 1);

  // ---- hconv2 + gnorm2 + fc2
  gemm(xl, nullptr, hb, l2t, nullptr, nullptr, Nn, Ff, Ff, 0);
  gemm(ea, nullptr, eb, l2t, nullptr, nullptr, Mm, Ff, Ff, 0);
  hconv_scatter(att2, bias2, xl, ea, hbuf);
  gnorm(hbuf, gn2_w, gn2_b, gn2_ms, h2b);
  gemm(out2, nullptr, h2b, fc2t, fc2_b, nullptr, Nn, HIDD, Ff, 1);

  // ---- feature attention: t = relu(out.T @ at1_w + at1_b); s = sigmoid(t @ at2_w + b) - mean
  const size_t TO = (size_t)FH2 * Nn;
  k_build_outT<<<dim3((unsigned)((TO + 255) / 256)), 256, 0, stream>>>(outTb, x, out1, out2, TO);
  gemm(tmat, nullptr, outTb, at1t, at1_b, nullptr, FH2, Nn, Nn, 2);
  k_rowdot<<<FH2, 256, 0, stream>>>(svec, tmat, at2_w, Nn, at2_b);
  k_sigmoid<<<dim3((FH2 + 255) / 256), 256, 0, stream>>>(svec, FH2);
  k_meansub<<<1, 256, 0, stream>>>(svec, FH2);

  // ---- classifier head
  const size_t HB = (size_t)BS * FH2;
  k_build_Hb0<<<dim3((unsigned)((HB + 255) / 256)), 256, 0, stream>>>(Hb0b, x, out1, out2, svec, HB);
  gemm(Hb1, Hb1b, Hb0b, cf1t, cf1_b, nullptr, BS, HIDD, FH2, 1);
  float* dout = (float*)d_out;
  k_head<<<BS, 256, 0, stream>>>(dout + (size_t)BS * 2, Hb1, dh_w, dh_b, HIDD);   // distil_logits
  gemm(Hb2, nullptr, Hb1b, cf2t, cf2_b, Hb1, BS, HIDD, HIDD, 1);                  // residual add
  k_head<<<BS, 256, 0, stream>>>(dout, Hb2, ch_w, ch_b, HIDD);                    // logits
  (void)n_in; (void)ws_size;
}
